// JointModalityAttention_12790412607628
// MI455X (gfx1250) — matrix operand order repaired and checked by reading, not compiled, not run on hardware
//
#include <hip/hip_runtime.h>

// ---------------------------------------------------------------------------
// JointModalityAttention on MI455X (gfx1250, wave32, WMMA 16x16x32 f16->f32)
//
// Shapes: B=8, N=512, D=512, H=8, dh=64, concatenated keys 3N=1536.
// Pipeline:
//   (a) convert x -> f16; transpose Wqkv (scale folded into Q cols) and Wout
//   (b) WMMA GEMM (async-LDS double-buffered): QKV = x @ Wqkv
//   (c) flash attention (online softmax via DPP row reductions,
//       whole-row masking == uniform rows)
//   (d) WMMA GEMM: out = O @ Wout (fp32 output)
// ---------------------------------------------------------------------------

typedef __attribute__((ext_vector_type(16))) _Float16 v16h;
typedef __attribute__((ext_vector_type(8)))  float    v8f;

union HalfVec { v16h v; uint4 q[2]; };
union Chunk8  { uint4 q; _Float16 h[8]; };

#define SCALE_Q 0.125f   // 64^-0.5

// ---- CDNA5 async global->LDS copy (ASYNCcnt) with synchronous fallback -----

#if defined(__gfx1250__) && __has_builtin(__builtin_amdgcn_global_load_async_to_lds_b128)
#define ASYNC_COPY 1
#endif

#ifdef ASYNC_COPY
// builtin signature (from toolchain diagnostic): params are pointers to
// "int __attribute__((vector_size(16)))" in AS1 (global) / AS3 (LDS).
typedef int v4i __attribute__((vector_size(16)));
typedef __attribute__((address_space(1))) v4i gv4i;
typedef __attribute__((address_space(3))) v4i sv4i;
#endif

__device__ inline void copy_b128_to_lds(_Float16* lds_dst, const _Float16* gsrc) {
#ifdef ASYNC_COPY
  __builtin_amdgcn_global_load_async_to_lds_b128(
      (gv4i*)gsrc, (sv4i*)lds_dst, 0, 0);
#else
  *(uint4*)lds_dst = *(const uint4*)gsrc;
#endif
}

template <int N>
__device__ inline void wait_async() {
#ifdef ASYNC_COPY
#if __has_builtin(__builtin_amdgcn_s_wait_asynccnt)
  __builtin_amdgcn_s_wait_asynccnt(N);
#else
  asm volatile("s_wait_asynccnt %0" ::"i"(N) : "memory");
#endif
#endif
}

// ---- wave32 16-lane row reductions via DPP (no DS-port traffic) -------------

template <int CTRL>
__device__ inline float dpp_xchg(float x) {
  return __builtin_bit_cast(
      float, __builtin_amdgcn_update_dpp(0, __builtin_bit_cast(int, x), CTRL,
                                         0xF, 0xF, true));
}
__device__ inline float row16_max(float x) {
  x = fmaxf(x, dpp_xchg<0xB1>(x));   // quad_perm [1,0,3,2]  (xor 1)
  x = fmaxf(x, dpp_xchg<0x4E>(x));   // quad_perm [2,3,0,1]  (xor 2)
  x = fmaxf(x, dpp_xchg<0x141>(x));  // row_half_mirror      (xor 7)
  x = fmaxf(x, dpp_xchg<0x140>(x));  // row_mirror           (xor 15)
  return x;
}
__device__ inline float row16_sum(float x) {
  x += dpp_xchg<0xB1>(x);
  x += dpp_xchg<0x4E>(x);
  x += dpp_xchg<0x141>(x);
  x += dpp_xchg<0x140>(x);
  return x;
}

// --------------------------- stage (a): converts -----------------------------

__global__ void convert_x_kernel(const float* __restrict__ x, _Float16* __restrict__ xh) {
  int idx = (blockIdx.x * 256 + threadIdx.x) * 4;
  float4 f = *(const float4*)(x + idx);
  xh[idx + 0] = (_Float16)f.x;
  xh[idx + 1] = (_Float16)f.y;
  xh[idx + 2] = (_Float16)f.z;
  xh[idx + 3] = (_Float16)f.w;
}

// Wqkv: [512][1536] row-major -> WT [1536][512] f16; fold softmax scale into Q cols.
__global__ void transpose_wqkv_kernel(const float* __restrict__ W, _Float16* __restrict__ WT) {
  int idx = blockIdx.x * 256 + threadIdx.x;      // 512*1536
  int d = idx / 1536, c = idx % 1536;
  float v = W[idx];
  if (c < 512) v *= SCALE_Q;
  WT[c * 512 + d] = (_Float16)v;
}

// Wout: [512][512] -> WT [512][512] transposed f16.
__global__ void transpose_wout_kernel(const float* __restrict__ W, _Float16* __restrict__ WT) {
  int idx = blockIdx.x * 256 + threadIdx.x;      // 512*512
  int d = idx / 512, c = idx % 512;
  WT[c * 512 + d] = (_Float16)W[idx];
}

// ------------------- stages (b)/(d): tiled WMMA GEMM -------------------------
// C[m,c] = sum_k A[m,k] * BT[c,k],  M=512, K=512, Ncols=1536 (QKV) or 512 (out).
// Block tile 128x128, K-step 32; 8 waves in 2(M)x4(N); per-wave 64x32 = 8 WMMAs.
// Double-buffered: async-copy tile kt+1 into buf^1 while WMMAs consume buf;
// s_wait_asynccnt 4 keeps the next tile's 4 copies/thread in flight.
// MODE 0: scatter-store f16 into Q / Kconcat / Vconcat attention layouts.
// MODE 1: store fp32 to d_out.

template <int MODE>
__global__ __launch_bounds__(256)
void wmma_gemm_kernel(const _Float16* __restrict__ Abase,   // [24][512][512]
                      const _Float16* __restrict__ BTbase,  // [3][Ncols][512]
                      int Ncols,
                      _Float16* __restrict__ Qh,            // [3][8][8][512][64]
                      _Float16* __restrict__ Kc,            // [8][8][1536][64]
                      _Float16* __restrict__ Vc,            // [8][8][1536][64]
                      float* __restrict__ Out)              // [3][8][512][512]
{
  constexpr int AST = 40;   // 32 halves + pad (80B rows, 16B aligned)
  constexpr int BST = 40;
  __shared__ alignas(16) _Float16 As[2][128 * AST];
  __shared__ alignas(16) _Float16 Bs[2][128 * BST];

  const int tid  = threadIdx.x;
  const int lane = tid & 31;
  const int w    = tid >> 5;
  const int wm   = w & 1, wn = w >> 1;
  const int ln   = lane & 15, g = lane >> 4;

  const int bz = blockIdx.z;            // i*8 + b
  const int i  = bz >> 3, b = bz & 7;
  const int c0 = blockIdx.x * 128;
  const int m0 = blockIdx.y * 128;

  const _Float16* A  = Abase  + (size_t)bz * (512 * 512);
  const _Float16* BT = BTbase + (size_t)i * Ncols * 512;

  const v8f vz = {0.f, 0.f, 0.f, 0.f, 0.f, 0.f, 0.f, 0.f};
  v8f acc[4][2];
  for (int mt = 0; mt < 4; ++mt)
    for (int nt = 0; nt < 2; ++nt) acc[mt][nt] = vz;

  // cooperative staging of one 128x32 K-tile of A and of BT (4 b128/thread)
  auto stage = [&](int buf, int kt) {
    const int k0 = kt * 32;
    for (int c = tid; c < 512; c += 256) {
      int row = c >> 2, part = c & 3;
      copy_b128_to_lds(&As[buf][row * AST + part * 8],
                       A + (size_t)(m0 + row) * 512 + k0 + part * 8);
      copy_b128_to_lds(&Bs[buf][row * BST + part * 8],
                       BT + (size_t)(c0 + row) * 512 + k0 + part * 8);
    }
  };

  stage(0, 0);
  for (int kt = 0; kt < 16; ++kt) {
    const int buf = kt & 1;
    if (kt + 1 < 16) {
      stage(buf ^ 1, kt + 1);
      wait_async<4>();     // previous tile resident; next 4 copies in flight
    } else {
      wait_async<0>();
    }
    __syncthreads();

    const _Float16* Asb = &As[buf][0];
    const _Float16* Bsb = &Bs[buf][0];

    // A fragments (16x32): lane = row m, halves k = g*8+{0..7} and 16+g*8+{0..7}
    v16h af[4], bf[2];
    #pragma unroll
    for (int mt = 0; mt < 4; ++mt) {
      const _Float16* p = Asb + (wm * 64 + mt * 16 + ln) * AST;
      HalfVec u;
      u.q[0] = *(const uint4*)(p + g * 8);
      u.q[1] = *(const uint4*)(p + 16 + g * 8);
      af[mt] = u.v;
    }
    // B fragments (32x16): lane = col n, halves k = g*8+{0..7} and 16+g*8+{0..7}
    #pragma unroll
    for (int nt = 0; nt < 2; ++nt) {
      const _Float16* p = Bsb + (wn * 32 + nt * 16 + ln) * BST;
      HalfVec u;
      u.q[0] = *(const uint4*)(p + g * 8);
      u.q[1] = *(const uint4*)(p + 16 + g * 8);
      bf[nt] = u.v;
    }
    #pragma unroll
    for (int mt = 0; mt < 4; ++mt)
      #pragma unroll
      for (int nt = 0; nt < 2; ++nt)
        acc[mt][nt] = __builtin_amdgcn_wmma_f32_16x16x32_f16(
            false, af[mt], false, bf[nt], (short)0, acc[mt][nt], false, false);
    __syncthreads();
  }

  // C layout: lane ln = col, VGPR v -> row v + 8*g within 16x16 tile
  if (MODE == 0) {
    #pragma unroll
    for (int mt = 0; mt < 4; ++mt)
      #pragma unroll
      for (int nt = 0; nt < 2; ++nt) {
        int col  = c0 + wn * 32 + nt * 16 + ln;
        int part = col >> 9, cc = col & 511;
        int head = cc >> 6, d = cc & 63;
        #pragma unroll
        for (int v = 0; v < 8; ++v) {
          int row = m0 + wm * 64 + mt * 16 + v + 8 * g;
          _Float16 hv = (_Float16)acc[mt][nt][v];
          if (part == 0)
            Qh[((((size_t)i * 8 + b) * 8 + head) * 512 + row) * 64 + d] = hv;
          else if (part == 1)
            Kc[(((size_t)b * 8 + head) * 1536 + i * 512 + row) * 64 + d] = hv;
          else
            Vc[(((size_t)b * 8 + head) * 1536 + i * 512 + row) * 64 + d] = hv;
        }
      }
  } else {
    #pragma unroll
    for (int mt = 0; mt < 4; ++mt)
      #pragma unroll
      for (int nt = 0; nt < 2; ++nt) {
        int col = c0 + wn * 32 + nt * 16 + ln;
        #pragma unroll
        for (int v = 0; v < 8; ++v) {
          int row = m0 + wm * 64 + mt * 16 + v + 8 * g;
          Out[(size_t)bz * (512 * 512) + (size_t)row * 512 + col] = acc[mt][nt][v];
        }
      }
  }
}

// ---------------------- stage (c): flash attention ---------------------------
// Block: 256 threads (8 waves), each wave owns 16 query rows; blockIdx.x tiles
// 128 query rows, blockIdx.y = i*64 + b*8 + h. Key blocks of 128 (12 iters).
// K staged via async-LDS copies; V staged transposed manually (needs VGPRs
// anyway); softmax row stats reduced with DPP inside each 16-lane row.

__global__ __launch_bounds__(256)
void flash_attn_kernel(const _Float16* __restrict__ Qh,
                       const _Float16* __restrict__ Kc,
                       const _Float16* __restrict__ Vc,
                       const int* __restrict__ mk0,
                       const int* __restrict__ mk1,
                       const int* __restrict__ mk2,
                       _Float16* __restrict__ Oh)   // [3][8][512][512] (n, h*64+d)
{
  constexpr int KST = 72;    // K rows: 64 halves + pad
  constexpr int VST = 136;   // V^T rows: 128 halves + pad
  constexpr int PST = 136;
  __shared__ alignas(16) _Float16 Ks[128 * KST];       // [j][d]
  __shared__ alignas(16) _Float16 Vs[64 * VST];        // transposed [d][j]
  __shared__ alignas(16) _Float16 Ps[8 * 16 * PST];    // per-wave P tile [16][128]

  const int tid = threadIdx.x, lane = tid & 31, w = tid >> 5;
  const int ln = lane & 15, g = lane >> 4;

  const int z = blockIdx.y;                 // i*64 + b*8 + h
  const int i = z >> 6, b = (z >> 3) & 7, h = z & 7;
  const int qrow0 = blockIdx.x * 128 + w * 16;

  const _Float16* Q = Qh + ((((size_t)i * 8 + b) * 8 + h) * 512) * 64;
  const _Float16* K = Kc + (((size_t)b * 8 + h) * 1536) * 64;
  const _Float16* V = Vc + (((size_t)b * 8 + h) * 1536) * 64;
  const int* mask = (i == 0) ? mk0 : (i == 1 ? mk1 : mk2);

  // Q fragments, dh=64 -> two 16x32 A-fragments, loaded once from global
  v16h aq[2];
  {
    const _Float16* qp = Q + (size_t)(qrow0 + ln) * 64;
    #pragma unroll
    for (int s = 0; s < 2; ++s) {
      HalfVec u;
      u.q[0] = *(const uint4*)(qp + s * 32 + g * 8);
      u.q[1] = *(const uint4*)(qp + s * 32 + 16 + g * 8);
      aq[s] = u.v;
    }
  }

  const v8f vz = {0.f, 0.f, 0.f, 0.f, 0.f, 0.f, 0.f, 0.f};
  v8f acc_o[4];
  #pragma unroll
  for (int t = 0; t < 4; ++t) acc_o[t] = vz;

  bool  rm[8];
  float mrow[8], lrow[8];
  #pragma unroll
  for (int v = 0; v < 8; ++v) {
    rm[v]   = mask[b * 512 + qrow0 + v + 8 * g] != 0;
    mrow[v] = -3.0e38f;
    lrow[v] = 0.f;
  }

  _Float16* Pw = &Ps[w * 16 * PST];

  for (int kb = 0; kb < 12; ++kb) {
    const int j0 = kb * 128;
    // K: async memory->LDS copies (ASYNCcnt), no VGPR round-trip
    for (int c = tid; c < 1024; c += 256) {
      int j = c >> 3, part = c & 7;
      copy_b128_to_lds(&Ks[j * KST + part * 8],
                       K + (size_t)(j0 + j) * 64 + part * 8);
    }
    // V: manual transposed staging (overlaps with the K copies in flight)
    for (int c = tid; c < 1024; c += 256) {
      int j = c >> 3, part = c & 7;
      Chunk8 u;
      u.q = *(const uint4*)(V + (size_t)(j0 + j) * 64 + part * 8);
      #pragma unroll
      for (int e = 0; e < 8; ++e) Vs[(part * 8 + e) * VST + j] = u.h[e];
    }
    wait_async<0>();
    __syncthreads();

    // S = Q K^T : 16 queries x 128 keys = 8 n-tiles x 2 k-steps
    v8f s_acc[8];
    #pragma unroll
    for (int t = 0; t < 8; ++t) s_acc[t] = vz;
    #pragma unroll
    for (int s = 0; s < 2; ++s) {
      #pragma unroll
      for (int t = 0; t < 8; ++t) {
        const _Float16* p = &Ks[(t * 16 + ln) * KST + s * 32 + g * 8];
        HalfVec u;
        u.q[0] = *(const uint4*)p;
        u.q[1] = *(const uint4*)(p + 16);
        s_acc[t] = __builtin_amdgcn_wmma_f32_16x16x32_f16(
            false, aq[s], false, u.v, (short)0, s_acc[t], false, false);
      }
    }

    // masked query rows: constant logits -> uniform softmax (== -1e9 fill)
    #pragma unroll
    for (int v = 0; v < 8; ++v)
      if (!rm[v]) {
        #pragma unroll
        for (int t = 0; t < 8; ++t) s_acc[t][v] = 0.f;
      }

    // online softmax; row stats reduced across each 16-lane row via DPP
    #pragma unroll
    for (int v = 0; v < 8; ++v) {
      float mx = s_acc[0][v];
      #pragma unroll
      for (int t = 1; t < 8; ++t) mx = fmaxf(mx, s_acc[t][v]);
      mx = row16_max(mx);
      float mnew = fmaxf(mrow[v], mx);
      float sc   = __expf(mrow[v] - mnew);
      lrow[v] *= sc;
      #pragma unroll
      for (int t = 0; t < 4; ++t) acc_o[t][v] *= sc;
      mrow[v] = mnew;
    }

    // P = exp(S - m) -> per-wave LDS tile (C layout -> A layout transpose)
    #pragma unroll
    for (int v = 0; v < 8; ++v) {
      float psum = 0.f;
      #pragma unroll
      for (int t = 0; t < 8; ++t) {
        float pe = __expf(s_acc[t][v] - mrow[v]);
        psum += pe;
        Pw[(v + 8 * g) * PST + t * 16 + ln] = (_Float16)pe;
      }
      lrow[v] += row16_sum(psum);
    }

    // O += P V : contraction over 128 keys = 4 k-steps x 4 d-tiles
    #pragma unroll
    for (int s2 = 0; s2 < 4; ++s2) {
      const _Float16* pp = &Pw[ln * PST + s2 * 32 + g * 8];
      HalfVec ua;
      ua.q[0] = *(const uint4*)pp;
      ua.q[1] = *(const uint4*)(pp + 16);
      #pragma unroll
      for (int t2 = 0; t2 < 4; ++t2) {
        const _Float16* vp = &Vs[(t2 * 16 + ln) * VST + s2 * 32 + g * 8];
        HalfVec ub;
        ub.q[0] = *(const uint4*)vp;
        ub.q[1] = *(const uint4*)(vp + 16);
        acc_o[t2] = __builtin_amdgcn_wmma_f32_16x16x32_f16(
            false, ua.v, false, ub.v, (short)0, acc_o[t2], false, false);
      }
    }
    __syncthreads();
  }

  // epilogue: O /= l, store f16 row-major [n][h*64+d]
  #pragma unroll
  for (int t2 = 0; t2 < 4; ++t2) {
    #pragma unroll
    for (int v = 0; v < 8; ++v) {
      float o  = acc_o[t2][v] / lrow[v];
      int  row = qrow0 + v + 8 * g;
      Oh[(((size_t)i * 8 + b) * 512 + row) * 512 + h * 64 + t2 * 16 + ln] = (_Float16)o;
    }
  }
}

// ------------------------------- launcher ------------------------------------

extern "C" void kernel_launch(void* const* d_in, const int* in_sizes, int n_in,
                              void* d_out, int out_size, void* d_ws, size_t ws_size,
                              hipStream_t stream) {
  // setup_inputs() dict order: x0,m0,Wqkv0,Wout0, x1,m1,Wqkv1,Wout1, x2,...
  const float* x[3]    = {(const float*)d_in[0], (const float*)d_in[4], (const float*)d_in[8]};
  const int*   m[3]    = {(const int*)d_in[1],   (const int*)d_in[5],   (const int*)d_in[9]};
  const float* wqkv[3] = {(const float*)d_in[2], (const float*)d_in[6], (const float*)d_in[10]};
  const float* wout[3] = {(const float*)d_in[3], (const float*)d_in[7], (const float*)d_in[11]};
  float* out = (float*)d_out;

  char* ws = (char*)d_ws;
  size_t off = 0;
  auto alloc = [&](size_t bytes) {
    void* p = ws + off;
    off += (bytes + 255) & ~(size_t)255;
    return p;
  };
  _Float16* xh    = (_Float16*)alloc((size_t)3 * 8 * 512 * 512 * 2);   // x in f16
  _Float16* wqkvT = (_Float16*)alloc((size_t)3 * 1536 * 512 * 2);      // Wqkv^T f16 (Q scaled)
  _Float16* woutT = (_Float16*)alloc((size_t)3 * 512 * 512 * 2);       // Wout^T f16
  _Float16* Qh    = (_Float16*)alloc((size_t)3 * 8 * 8 * 512 * 64 * 2);
  _Float16* Kc    = (_Float16*)alloc((size_t)8 * 8 * 1536 * 64 * 2);
  _Float16* Vc    = (_Float16*)alloc((size_t)8 * 8 * 1536 * 64 * 2);
  _Float16* Oh    = (_Float16*)alloc((size_t)3 * 8 * 512 * 512 * 2);

  for (int i = 0; i < 3; ++i) {
    convert_x_kernel<<<2048, 256, 0, stream>>>(x[i], xh + (size_t)i * 8 * 512 * 512);
    transpose_wqkv_kernel<<<3072, 256, 0, stream>>>(wqkv[i], wqkvT + (size_t)i * 1536 * 512);
    transpose_wout_kernel<<<1024, 256, 0, stream>>>(wout[i], woutT + (size_t)i * 512 * 512);
  }

  dim3 g1(12, 4, 24);   // 1536/128 cols, 512/128 rows, 3 modalities * 8 batch
  wmma_gemm_kernel<0><<<g1, 256, 0, stream>>>(xh, wqkvT, 1536, Qh, Kc, Vc, nullptr);

  dim3 g2(4, 192);      // 512/128 query tiles, 3*8*8 (i,b,h)
  flash_attn_kernel<<<g2, 256, 0, stream>>>(Qh, Kc, Vc, m[0], m[1], m[2], Oh);

  dim3 g3(4, 4, 24);    // 512/128 cols, 512/128 rows, 24
  wmma_gemm_kernel<1><<<g3, 256, 0, stream>>>(Oh, woutT, 512, nullptr, nullptr, nullptr, out);
}
